// MultiHeadAttention_70643622084877
// MI455X (gfx1250) — compile-verified, matrix operand order checked by reading, not run on hardware
//
#include <hip/hip_runtime.h>
#include <hip/hip_bf16.h>

// ---------------------------------------------------------------------------
// MHA forward for MI455X (gfx1250, wave32).  All GEMMs via
// v_wmma_f32_16x16x32_bf16.  Softmax in f32.  attn_proba written once,
// consumed from LDS (never re-read from HBM).
// ---------------------------------------------------------------------------

#define D_MODEL 1024
#define SEQ     1024
#define BATCH   4
#define HEADS   16
#define DHEAD   64
#define MROWS   (BATCH * SEQ)   // 4096

typedef __attribute__((ext_vector_type(16))) __bf16 v16bf;
typedef __attribute__((ext_vector_type(8)))  __bf16 v8bf;
typedef __attribute__((ext_vector_type(4)))  __bf16 v4bf;
typedef __attribute__((ext_vector_type(8)))  float  v8f;
typedef __attribute__((ext_vector_type(4)))  float  v4f;

// Load one wave-fragment (A or B) of a 16x32 bf16 tile.
// rowptr = start of this lane's row (M for A, N for B), row-major, K-contig.
// Per ISA 16-bit layout: lanes 0-15 hold K {k0..k0+7, k0+16..k0+23},
// lanes 16-31 hold K {k0+8..k0+15, k0+24..k0+31}.  hf = lane>>4.
__device__ __forceinline__ v16bf load_frag(const __bf16* rowptr, int k0, int hf) {
  union { v16bf v; v8bf h[2]; } u;
  const __bf16* p = rowptr + k0 + hf * 8;
  u.h[0] = *reinterpret_cast<const v8bf*>(p);
  u.h[1] = *reinterpret_cast<const v8bf*>(p + 16);
  return u.v;
}

__device__ __forceinline__ v8f wmma_bf16(v16bf a, v16bf b, v8f c) {
  return __builtin_amdgcn_wmma_f32_16x16x32_bf16(
      /*neg_a=*/false, a, /*neg_b=*/false, b,
      /*c_mod=*/(short)0, c, /*reuse_a=*/false, /*reuse_b=*/false);
}

// ---------------------------------------------------------------------------
// f32 -> bf16 conversion (vectorized x4)
// ---------------------------------------------------------------------------
__global__ void cvt_f32_to_bf16(const float* __restrict__ in,
                                __bf16* __restrict__ out, int n4) {
  int i = blockIdx.x * blockDim.x + threadIdx.x;
  if (i < n4) {
    v4f x = *reinterpret_cast<const v4f*>(in + (size_t)i * 4);
    v4bf o;
    o.x = (__bf16)x.x; o.y = (__bf16)x.y; o.z = (__bf16)x.z; o.w = (__bf16)x.w;
    *reinterpret_cast<v4bf*>(out + (size_t)i * 4) = o;
  }
}

// ---------------------------------------------------------------------------
// Projection: out[m,n] = sum_k X[m,k] * W[n,k] + bias[n]   (torch Linear)
// X: [4096,1024] bf16 row-major, W: [1024,1024] bf16 row-major.
// Stores bf16 into head-major layouts:
//   layout 0:  [b,h,s,d]  (Q, K)
//   layout 1:  [b,h,d,s]  (V transposed, so P@V B-fragments are contiguous)
// Block: 128 threads (4 waves).  Wave tile: 16x64 (A reused over 4 N-tiles).
// Grid: (4096/16, 1024/256) = (256, 4)
// ---------------------------------------------------------------------------
__global__ void proj_kernel(const __bf16* __restrict__ X,
                            const __bf16* __restrict__ W,
                            const float* __restrict__ bias,
                            __bf16* __restrict__ out, int layout) {
  const int tid  = threadIdx.x;
  const int wave = tid >> 5;
  const int lane = tid & 31;
  const int hf   = lane >> 4;
  const int r    = lane & 15;
  const int m0   = blockIdx.x * 16;
  const int n0   = blockIdx.y * 256 + wave * 64;

  const __bf16* arow = X + (size_t)(m0 + r) * D_MODEL;
  v8f acc[4] = {};

  for (int k0 = 0; k0 < D_MODEL; k0 += 32) {
    v16bf a = load_frag(arow, k0, hf);
#pragma unroll
    for (int j = 0; j < 4; ++j) {
      v16bf b = load_frag(W + (size_t)(n0 + j * 16 + r) * D_MODEL, k0, hf);
      acc[j] = wmma_bf16(a, b, acc[j]);
    }
  }

#pragma unroll
  for (int j = 0; j < 4; ++j) {
#pragma unroll
    for (int e = 0; e < 8; ++e) {
      int m = m0 + e + hf * 8;          // C layout: VGPR e, halves are M / M+8
      int n = n0 + j * 16 + r;
      float val = acc[j][e] + bias[n];
      int b = m >> 10, s = m & 1023;
      int h = n >> 6,  d = n & 63;
      size_t idx;
      if (layout == 0)
        idx = (((size_t)(b * HEADS + h) * SEQ + s) * DHEAD) + d;
      else
        idx = (((size_t)(b * HEADS + h) * DHEAD + d) * SEQ) + s;
      out[idx] = (__bf16)val;
    }
  }
}

// ---------------------------------------------------------------------------
// Output projection: f32 result into d_out[0 .. 4M)
// ---------------------------------------------------------------------------
__global__ void out_proj_kernel(const __bf16* __restrict__ X,
                                const __bf16* __restrict__ W,
                                const float* __restrict__ bias,
                                float* __restrict__ out) {
  const int tid  = threadIdx.x;
  const int wave = tid >> 5;
  const int lane = tid & 31;
  const int hf   = lane >> 4;
  const int r    = lane & 15;
  const int m0   = blockIdx.x * 16;
  const int n0   = blockIdx.y * 256 + wave * 64;

  const __bf16* arow = X + (size_t)(m0 + r) * D_MODEL;
  v8f acc[4] = {};

  for (int k0 = 0; k0 < D_MODEL; k0 += 32) {
    v16bf a = load_frag(arow, k0, hf);
#pragma unroll
    for (int j = 0; j < 4; ++j) {
      v16bf b = load_frag(W + (size_t)(n0 + j * 16 + r) * D_MODEL, k0, hf);
      acc[j] = wmma_bf16(a, b, acc[j]);
    }
  }

#pragma unroll
  for (int j = 0; j < 4; ++j) {
#pragma unroll
    for (int e = 0; e < 8; ++e) {
      int m = m0 + e + hf * 8;
      int n = n0 + j * 16 + r;
      out[(size_t)m * D_MODEL + n] = acc[j][e] + bias[n];
    }
  }
}

// ---------------------------------------------------------------------------
// Fused attention: one block (128 threads / 4 waves) per (b, h, 16-row qblock)
//   phase 1: scores 16x1024 via WMMA (K=64), scale, mask -> LDS f32
//   phase 2: softmax (row max / sum), write attn_proba f32, pack bf16 P
//   phase 3: P(16x1024) @ V(1024x64) via WMMA -> bf16 context [b,s,h*64+d]
// LDS: 64K scores + 33K padded P + reductions (~97.3 KB of 320 KB WGP LDS).
// ---------------------------------------------------------------------------
#define P_LD 1032   // padded bf16 row stride (2064 B, 16B-aligned, no bank conflicts)

__global__ void attn_kernel(const __bf16* __restrict__ Qh,
                            const __bf16* __restrict__ Kh,
                            const __bf16* __restrict__ Vt,
                            const int* __restrict__ mask,
                            float* __restrict__ attn_out,
                            __bf16* __restrict__ Oc) {
  extern __shared__ char smem[];
  float*  s_scores = (float*)smem;                              // 16*1024 f32
  __bf16* s_p      = (__bf16*)(smem + 16 * 1024 * 4);           // 16*P_LD bf16
  float*  s_redm   = (float*)(smem + 16 * 1024 * 4 + 16 * P_LD * 2); // 16*8
  float*  s_reds   = s_redm + 128;                              // 16*8
  float*  s_inv    = s_reds + 128;                              // 16

  const int tid  = threadIdx.x;
  const int wave = tid >> 5;
  const int lane = tid & 31;
  const int hf   = lane >> 4;
  const int r    = lane & 15;

  const int blk = blockIdx.x;            // 4096 blocks
  const int b   = blk >> 10;
  const int h   = (blk >> 6) & 15;
  const int q0  = (blk & 63) * 16;

  const size_t bh = (size_t)(b * HEADS + h);
  const __bf16* Qbase = Qh + (bh * SEQ + q0) * DHEAD;
  const __bf16* Kbase = Kh + bh * SEQ * DHEAD;
  const __bf16* Vbase = Vt + bh * DHEAD * SEQ;
  const int*    mbase = mask + (size_t)b * SEQ * SEQ;

  // ---- phase 1: scores = (Q K^T) / 8, masked, into LDS -------------------
  const __bf16* qrow = Qbase + (size_t)r * DHEAD;
  v16bf a0 = load_frag(qrow, 0, hf);
  v16bf a1 = load_frag(qrow, 32, hf);

  for (int t = 0; t < 16; ++t) {
    const int c0 = wave * 256 + t * 16;
    const __bf16* krow = Kbase + (size_t)(c0 + r) * DHEAD;
    v16bf b0 = load_frag(krow, 0, hf);
    v16bf b1 = load_frag(krow, 32, hf);
    v8f c = {};
    c = wmma_bf16(a0, b0, c);
    c = wmma_bf16(a1, b1, c);
#pragma unroll
    for (int e = 0; e < 8; ++e) {
      const int row = e + hf * 8;
      const int col = c0 + r;
      float val = c[e] * 0.125f;                        // 1/sqrt(64)
      if (mbase[(size_t)(q0 + row) * SEQ + col] == 1)
        val = -1e-9f;                                   // faithful to reference
      s_scores[row * 1024 + col] = val;
    }
  }
  __syncthreads();

  // ---- phase 2: softmax over each row ------------------------------------
  {
    const int row = tid >> 3;   // 16 rows, 8 threads/row
    const int seg = tid & 7;
    float* sr = s_scores + row * 1024;
    float mx = -INFINITY;
    for (int i = 0; i < 128; ++i) mx = fmaxf(mx, sr[seg + 8 * i]);
    s_redm[row * 8 + seg] = mx;
    __syncthreads();
    float rm = s_redm[row * 8];
#pragma unroll
    for (int i = 1; i < 8; ++i) rm = fmaxf(rm, s_redm[row * 8 + i]);
    float sum = 0.f;
    for (int i = 0; i < 128; ++i) {
      const int c = seg + 8 * i;
      float e = __expf(sr[c] - rm);
      sr[c] = e;
      sum += e;
    }
    s_reds[row * 8 + seg] = sum;
    __syncthreads();
    if (seg == 0) {
      float s = 0.f;
#pragma unroll
      for (int i = 0; i < 8; ++i) s += s_reds[row * 8 + i];
      s_inv[row] = 1.0f / s;
    }
    __syncthreads();
  }

  // ---- normalize: write attn_proba (f32, coalesced) + pack bf16 P --------
  float* attn_g = attn_out + (bh * SEQ + q0) * SEQ;
  for (int idx4 = tid; idx4 < 16 * 256; idx4 += 128) {
    const int row = idx4 >> 8;
    const int c   = (idx4 & 255) * 4;
    const float inv = s_inv[row];
    v4f p = *reinterpret_cast<v4f*>(s_scores + row * 1024 + c);
    p.x *= inv; p.y *= inv; p.z *= inv; p.w *= inv;
    *reinterpret_cast<v4f*>(attn_g + (size_t)row * SEQ + c) = p;
    v4bf pb;
    pb.x = (__bf16)p.x; pb.y = (__bf16)p.y; pb.z = (__bf16)p.z; pb.w = (__bf16)p.w;
    *reinterpret_cast<v4bf*>(s_p + row * P_LD + c) = pb;
  }
  __syncthreads();

  // ---- phase 3: context = P @ V, one 16x16 tile per wave (Dh = 4*16) -----
  {
    const int d0 = wave * 16;
    const __bf16* prow = s_p + (size_t)r * P_LD;
    const __bf16* vrow = Vbase + (size_t)(d0 + r) * SEQ;
    v8f c = {};
    for (int k0 = 0; k0 < SEQ; k0 += 32) {
      v16bf a  = load_frag(prow, k0, hf);
      v16bf bb = load_frag(vrow, k0, hf);
      c = wmma_bf16(a, bb, c);
    }
#pragma unroll
    for (int e = 0; e < 8; ++e) {
      const int row = e + hf * 8;
      const int col = d0 + r;
      Oc[((size_t)b * SEQ + q0 + row) * D_MODEL + h * DHEAD + col] = (__bf16)c[e];
    }
  }
}

// ---------------------------------------------------------------------------
// Launcher
// ---------------------------------------------------------------------------
extern "C" void kernel_launch(void* const* d_in, const int* in_sizes, int n_in,
                              void* d_out, int out_size, void* d_ws, size_t ws_size,
                              hipStream_t stream) {
  const float* q    = (const float*)d_in[0];
  const float* k    = (const float*)d_in[1];
  const float* v    = (const float*)d_in[2];
  const int*   mask = (const int*)d_in[3];
  const float* w_q  = (const float*)d_in[4];
  const float* b_q  = (const float*)d_in[5];
  const float* w_k  = (const float*)d_in[6];
  const float* b_k  = (const float*)d_in[7];
  const float* w_v  = (const float*)d_in[8];
  const float* b_v  = (const float*)d_in[9];
  const float* w_o  = (const float*)d_in[10];
  const float* b_o  = (const float*)d_in[11];

  float* out_main   = (float*)d_out;                                  // [4,1024,1024]
  float* attn_proba = out_main + (size_t)BATCH * SEQ * D_MODEL;       // [4,16,1024,1024]

  const size_t ACT = (size_t)MROWS * D_MODEL;     // 4M elems
  const size_t WEL = (size_t)D_MODEL * D_MODEL;   // 1M elems

  char* ws = (char*)d_ws;
  __bf16* qb  = (__bf16*)ws; ws += ACT * 2;
  __bf16* kb  = (__bf16*)ws; ws += ACT * 2;
  __bf16* vb  = (__bf16*)ws; ws += ACT * 2;
  __bf16* wqb = (__bf16*)ws; ws += WEL * 2;
  __bf16* wkb = (__bf16*)ws; ws += WEL * 2;
  __bf16* wvb = (__bf16*)ws; ws += WEL * 2;
  __bf16* wob = (__bf16*)ws; ws += WEL * 2;
  __bf16* Qh  = (__bf16*)ws; ws += ACT * 2;       // [b,h,s,d]
  __bf16* Kh  = (__bf16*)ws; ws += ACT * 2;       // [b,h,s,d]
  __bf16* Vt  = (__bf16*)ws; ws += ACT * 2;       // [b,h,d,s]
  __bf16* Oc  = (__bf16*)ws; ws += ACT * 2;       // [b,s,h*64+d]

  // 1) f32 -> bf16 conversions
  {
    const int actB = (int)(ACT / 4 + 255) / 256;
    const int wB   = (int)(WEL / 4 + 255) / 256;
    cvt_f32_to_bf16<<<actB, 256, 0, stream>>>(q, qb, (int)(ACT / 4));
    cvt_f32_to_bf16<<<actB, 256, 0, stream>>>(k, kb, (int)(ACT / 4));
    cvt_f32_to_bf16<<<actB, 256, 0, stream>>>(v, vb, (int)(ACT / 4));
    cvt_f32_to_bf16<<<wB, 256, 0, stream>>>(w_q, wqb, (int)(WEL / 4));
    cvt_f32_to_bf16<<<wB, 256, 0, stream>>>(w_k, wkb, (int)(WEL / 4));
    cvt_f32_to_bf16<<<wB, 256, 0, stream>>>(w_v, wvb, (int)(WEL / 4));
    cvt_f32_to_bf16<<<wB, 256, 0, stream>>>(w_o, wob, (int)(WEL / 4));
  }

  // 2) Q/K/V projections (WMMA)
  dim3 pgrid(MROWS / 16, D_MODEL / 256);
  proj_kernel<<<pgrid, 128, 0, stream>>>(qb, wqb, b_q, Qh, 0);
  proj_kernel<<<pgrid, 128, 0, stream>>>(kb, wkb, b_k, Kh, 0);
  proj_kernel<<<pgrid, 128, 0, stream>>>(vb, wvb, b_v, Vt, 1);

  // 3) fused attention (scores + softmax + attn_proba + P@V)
  const int smemBytes = 16 * 1024 * 4 + 16 * P_LD * 2 + 128 * 4 + 128 * 4 + 16 * 4;
  attn_kernel<<<BATCH * HEADS * (SEQ / 16), 128, smemBytes, stream>>>(
      Qh, Kh, Vt, mask, attn_proba, Oc);

  // 4) output projection (WMMA, f32 out)
  out_proj_kernel<<<pgrid, 128, 0, stream>>>(Oc, wob, b_o, out_main);
}
